// Net_47674136985816
// MI455X (gfx1250) — matrix operand order repaired and checked by reading, not compiled, not run on hardware
//
#include <hip/hip_runtime.h>
#include <hip/hip_bf16.h>

typedef __attribute__((ext_vector_type(16))) _Float16 v16h;
typedef __attribute__((ext_vector_type(8)))  float    v8f;

#define HID 32
#define FEAT 128

// ---------------------------------------------------------------------------
// init: deg = 1 (self loop), agg = 0
// ---------------------------------------------------------------------------
__global__ void init_kernel(float* __restrict__ deg, float* __restrict__ agg, int N) {
    int t = blockIdx.x * blockDim.x + threadIdx.x;
    if (t < N * HID) agg[t] = 0.0f;
    if (t < N) deg[t] = 1.0f;
}

// ---------------------------------------------------------------------------
// degree histogram over dst (edges are int64)
// ---------------------------------------------------------------------------
__global__ void deg_kernel(const long long* __restrict__ edst, float* __restrict__ deg, int E) {
    int e = blockIdx.x * blockDim.x + threadIdx.x;
    if (e >= E) return;
    atomicAdd(&deg[(int)edst[e]], 1.0f);
}

__global__ void dinv_kernel(float* __restrict__ deg, int N) {
    int i = blockIdx.x * blockDim.x + threadIdx.x;
    if (i >= N) return;
    deg[i] = rsqrtf(deg[i]);   // in place: deg -> deg^{-1/2}
}

// ---------------------------------------------------------------------------
// Pack W (K x 32, f32 row-major) into the WMMA B-operand lane layout, f16.
// t = (c * kblocks + kb) * 32 + lane ; element j of lane's v16h is
// K = kb*32 + (lane>=16 ? 8 : 0) + (j<8 ? j : 16 + (j-8)), N-col = c*16 + (lane&15).
// ---------------------------------------------------------------------------
__global__ void pack_w_f16(const float* __restrict__ W, _Float16* __restrict__ Bpk, int kblocks) {
    int t = blockIdx.x * blockDim.x + threadIdx.x;
    int total = 2 * kblocks * 32;
    if (t >= total) return;
    int lane    = t & 31;
    int kb      = (t >> 5) % kblocks;
    int c       = t / (kblocks * 32);
    int l16     = lane & 15;
    int halfsel = lane >> 4;
    int col     = c * 16 + l16;
    for (int j = 0; j < 16; ++j) {
        int k = kb * 32 + halfsel * 8 + (j < 8 ? j : 16 + (j - 8));
        Bpk[(size_t)t * 16 + j] = (_Float16)W[(size_t)k * HID + col];
    }
}

// ---------------------------------------------------------------------------
// H = X @ W using v_wmma_f32_16x16x32_f16.
// X: N x (kblocks*32) f32 row-major.  Bpk: packed f16 B operand (see above).
// One wave -> one 16-row tile, both 16-col tiles (2 WMMAs per K step).
// ---------------------------------------------------------------------------
__global__ void gemm_wmma_f16(const float* __restrict__ X, const _Float16* __restrict__ Bpk,
                              float* __restrict__ H, int n_tiles, int kblocks) {
    int gwave = (blockIdx.x * blockDim.x + threadIdx.x) >> 5;
    int lane  = threadIdx.x & 31;
    if (gwave >= n_tiles) return;                 // wave-uniform: EXEC stays all-ones

    int l16     = lane & 15;
    int halfsel = lane >> 4;
    int row     = gwave * 16 + l16;
    int ld      = kblocks * 32;
    const float* xrow = X + (size_t)row * ld + halfsel * 8;

    v8f acc0 = {};
    v8f acc1 = {};
    for (int kb = 0; kb < kblocks; ++kb) {
        const float* p = xrow + kb * 32;
        float4 a0 = *(const float4*)(p + 0);      // K = off+0..3
        float4 a1 = *(const float4*)(p + 4);      // K = off+4..7
        float4 a2 = *(const float4*)(p + 16);     // K = off+16..19
        float4 a3 = *(const float4*)(p + 20);     // K = off+20..23
        v16h a;
        a[0]  = (_Float16)a0.x; a[1]  = (_Float16)a0.y; a[2]  = (_Float16)a0.z; a[3]  = (_Float16)a0.w;
        a[4]  = (_Float16)a1.x; a[5]  = (_Float16)a1.y; a[6]  = (_Float16)a1.z; a[7]  = (_Float16)a1.w;
        a[8]  = (_Float16)a2.x; a[9]  = (_Float16)a2.y; a[10] = (_Float16)a2.z; a[11] = (_Float16)a2.w;
        a[12] = (_Float16)a3.x; a[13] = (_Float16)a3.y; a[14] = (_Float16)a3.z; a[15] = (_Float16)a3.w;

        v16h b0 = *(const v16h*)(Bpk + ((size_t)(0 * kblocks + kb) * 32 + lane) * 16);
        v16h b1 = *(const v16h*)(Bpk + ((size_t)(1 * kblocks + kb) * 32 + lane) * 16);

        acc0 = __builtin_amdgcn_wmma_f32_16x16x32_f16(false, a, false, b0, (short)0, acc0, false, false);
        acc1 = __builtin_amdgcn_wmma_f32_16x16x32_f16(false, a, false, b1, (short)0, acc1, false, false);
    }

    int baserow = gwave * 16 + halfsel * 8;       // C/D layout: VGPR r -> M = r (+8 for hi lanes)
    for (int r = 0; r < 8; ++r) {
        H[(size_t)(baserow + r) * HID + l16]      = acc0[r];
        H[(size_t)(baserow + r) * HID + 16 + l16] = acc1[r];
    }
}

// ---------------------------------------------------------------------------
// Edge scatter: agg[dst] += h[src] * dinv[src]*dinv[dst].  8 threads / edge,
// float4 gather + 4 global f32 atomics each.  Everything is L2-resident.
// ---------------------------------------------------------------------------
__global__ void scatter_kernel(const float* __restrict__ H,
                               const long long* __restrict__ esrc,
                               const long long* __restrict__ edst,
                               const float* __restrict__ dinv,
                               float* __restrict__ agg, int E) {
    long long t = (long long)blockIdx.x * blockDim.x + threadIdx.x;
    long long e = t >> 3;
    if (e >= E) return;
    int q = (int)(t & 7);
    int s = (int)esrc[e];
    int d = (int)edst[e];
    float norm = dinv[s] * dinv[d];
    float4 v = *(const float4*)(H + (size_t)s * HID + q * 4);
    float* ap = agg + (size_t)d * HID + q * 4;
    atomicAdd(ap + 0, v.x * norm);
    atomicAdd(ap + 1, v.y * norm);
    atomicAdd(ap + 2, v.z * norm);
    atomicAdd(ap + 3, v.w * norm);
}

// ---------------------------------------------------------------------------
// Epilogue: x = tanh(agg + dinv^2 * h + b); re-zero agg for the next pass.
// ---------------------------------------------------------------------------
__global__ void finish_kernel(float* __restrict__ agg, const float* __restrict__ H,
                              const float* __restrict__ dinv, const float* __restrict__ bias,
                              float* __restrict__ Xo, int N) {
    int t = blockIdx.x * blockDim.x + threadIdx.x;
    int n = t >> 3;
    if (n >= N) return;
    int q = t & 7;
    float di = dinv[n];
    float sw = di * di;                           // self-loop norm = 1/deg
    size_t off = (size_t)n * HID + q * 4;
    float4 av = *(const float4*)(agg + off);
    float4 hv = *(const float4*)(H + off);
    float4 bv = *(const float4*)(bias + q * 4);
    float4 o;
    o.x = tanhf(av.x + sw * hv.x + bv.x);
    o.y = tanhf(av.y + sw * hv.y + bv.y);
    o.z = tanhf(av.z + sw * hv.z + bv.z);
    o.w = tanhf(av.w + sw * hv.w + bv.w);
    *(float4*)(Xo + off) = o;
    float4 z = make_float4(0.f, 0.f, 0.f, 0.f);
    *(float4*)(agg + off) = z;
}

// ---------------------------------------------------------------------------
// threefry2x32 reproduction of jax.random.normal(jax.random.key(42), (N,1))
// ---------------------------------------------------------------------------
__device__ __forceinline__ unsigned rotl32(unsigned x, int r) {
    return (x << r) | (x >> (32 - r));
}

__device__ float jax_noise(unsigned i, unsigned half) {
    unsigned p  = (i < half) ? i : (i - half);
    unsigned x0 = p;
    unsigned x1 = p + half;
    const unsigned k0 = 0u, k1 = 42u, k2 = k0 ^ k1 ^ 0x1BD11BDAu;
    unsigned ks[3] = {k0, k1, k2};
    x0 += k0; x1 += k1;
    const int R[8] = {13, 15, 26, 6, 17, 29, 16, 24};
    for (int g = 0; g < 5; ++g) {
        const int* r = (g & 1) ? (R + 4) : R;
        for (int j = 0; j < 4; ++j) { x0 += x1; x1 = rotl32(x1, r[j]); x1 ^= x0; }
        x0 += ks[(g + 1) % 3];
        x1 += ks[(g + 2) % 3] + (unsigned)(g + 1);
    }
    unsigned bits = (i < half) ? x0 : x1;
    float f = __uint_as_float((bits >> 9) | 0x3f800000u) - 1.0f;   // [0,1)
    const float minv = -0.999999940395355224609375f;               // nextafter(-1,0)
    float u = fmaxf(minv, f * (1.0f - minv) + minv);
    return 1.4142135623730951f * erfinvf(u);
}

__global__ void output_kernel(const float* __restrict__ X, const float* __restrict__ Wout,
                              const float* __restrict__ bout, float* __restrict__ out, int N) {
    int n = blockIdx.x * blockDim.x + threadIdx.x;
    if (n >= N) return;
    const float* xr = X + (size_t)n * HID;
    float s = 0.0f;
    #pragma unroll
    for (int j = 0; j < HID; ++j) s += xr[j] * Wout[j];
    float v = tanhf(s + bout[0]);
    out[n] = v + jax_noise((unsigned)n, (unsigned)(N >> 1));
}

// ---------------------------------------------------------------------------
// Host-side launcher
// ---------------------------------------------------------------------------
extern "C" void kernel_launch(void* const* d_in, const int* in_sizes, int n_in,
                              void* d_out, int out_size, void* d_ws, size_t ws_size,
                              hipStream_t stream) {
    const float*     node_fea = (const float*)d_in[0];
    const float*     W_in     = (const float*)d_in[1];
    const float*     b_in     = (const float*)d_in[2];
    const float*     W_pass   = (const float*)d_in[3];
    const float*     b_pass   = (const float*)d_in[4];
    const float*     W_out    = (const float*)d_in[5];
    const float*     b_out    = (const float*)d_in[6];
    const long long* edges    = (const long long*)d_in[7];
    (void)n_in; (void)out_size; (void)ws_size;

    const int N = in_sizes[0] / FEAT;     // 100000
    const int E = in_sizes[7] / 2;        // 3200000
    const int KB_IN = FEAT / 32;          // 4
    const int KB_HID = HID / 32;          // 1

    // workspace carve-out (256B aligned)
    char* ws = (char*)d_ws;
    size_t o = 0;
    auto carve = [&](size_t bytes) -> char* {
        char* p = ws + o;
        o += (bytes + 255) & ~(size_t)255;
        return p;
    };
    float*     dinv    = (float*)    carve((size_t)N * 4);
    float*     h       = (float*)    carve((size_t)N * HID * 4);
    float*     agg     = (float*)    carve((size_t)N * HID * 4);
    float*     x       = (float*)    carve((size_t)N * HID * 4);
    _Float16*  BpkIn   = (_Float16*) carve((size_t)2 * KB_IN  * 32 * 16 * 2);
    _Float16*  BpkPass = (_Float16*) carve((size_t)2 * KB_HID * 32 * 16 * 2);

    const long long* esrc = edges;
    const long long* edst = edges + E;

    const int n_tiles = N / 16;                                 // 6250 (N % 16 == 0)
    const int gemm_blocks = (n_tiles + 7) / 8;                  // 8 waves / block
    const long long sthreads = (long long)E * 8;
    const int scatter_blocks = (int)((sthreads + 255) / 256);

    // --- normalization (recomputed every launch; ws is not persistent) ---
    init_kernel<<<(N * HID + 255) / 256, 256, 0, stream>>>(dinv, agg, N);
    deg_kernel<<<(E + 255) / 256, 256, 0, stream>>>(edst, dinv, E);
    dinv_kernel<<<(N + 255) / 256, 256, 0, stream>>>(dinv, N);

    // --- pack weights into WMMA B layout (f16) ---
    pack_w_f16<<<1, 256, 0, stream>>>(W_in,   BpkIn,   KB_IN);
    pack_w_f16<<<1, 64,  0, stream>>>(W_pass, BpkPass, KB_HID);

    // --- input layer: h = node_fea @ W_in ; conv ; tanh ---
    gemm_wmma_f16<<<gemm_blocks, 256, 0, stream>>>(node_fea, BpkIn, h, n_tiles, KB_IN);
    scatter_kernel<<<scatter_blocks, 256, 0, stream>>>(h, esrc, edst, dinv, agg, E);
    finish_kernel<<<(N * 8 + 255) / 256, 256, 0, stream>>>(agg, h, dinv, b_in, x, N);

    // --- 20 message-passing layers ---
    for (int p = 0; p < 20; ++p) {
        gemm_wmma_f16<<<gemm_blocks, 256, 0, stream>>>(x, BpkPass, h, n_tiles, KB_HID);
        scatter_kernel<<<scatter_blocks, 256, 0, stream>>>(h, esrc, edst, dinv, agg, E);
        finish_kernel<<<(N * 8 + 255) / 256, 256, 0, stream>>>(agg, h, dinv, b_pass, x, N);
    }

    // --- readout + threefry noise ---
    output_kernel<<<(N + 255) / 256, 256, 0, stream>>>(x, W_out, b_out, (float*)d_out, N);
}
